// VAEDMP_25950192402825
// MI455X (gfx1250) — matrix operand order repaired and checked by reading, not compiled, not run on hardware
//
#include <hip/hip_runtime.h>
#include <hip/hip_bf16.h>
#include <math.h>

// ---------------------------------------------------------------------------
// VAE-DMP on MI455X (gfx1250): all GEMMs via v_wmma_f32_16x16x32_bf16.
//  - Weights converted f32->bf16 once per launch (live in L2 afterwards).
//  - Double-buffered LDS pipeline: next tile staged while WMMAs run.
//  - Weight tile staged with CDNA5 async copy (global_load_async_to_lds_b128,
//    ASYNCcnt) -- bypasses VGPRs; activation tile staged synchronously since
//    it needs f32->bf16 VALU conversion.
//  - Batch-major activation layout => encoder/decoder write d_out directly.
// ---------------------------------------------------------------------------

typedef __attribute__((ext_vector_type(16))) __bf16 v16bf;
typedef __attribute__((ext_vector_type(8)))  __bf16 v8bf;
typedef __attribute__((ext_vector_type(8)))  float  v8f;
typedef __attribute__((ext_vector_type(4)))  float  v4f;

// Problem constants (match reference)
constexpr int   BD   = 256;   // batch
constexpr int   TD   = 128;   // time
constexpr int   XDIM = 512;
constexpr int   ZDIM = 256;
constexpr int   HID  = 1024;
constexpr int   TRANS= 512;
constexpr float C_ALPHA = 25.0f;
constexpr float C_BETA  = 6.25f;
constexpr float C_TAU   = 1.0f;
constexpr float C_DT    = 0.01f;
constexpr float C_SLOPE = 0.01f;
constexpr float C_A1 = 1.0f - C_DT * C_DT * C_ALPHA * C_BETA / C_TAU;
constexpr float C_A2 = (1.0f - C_DT * C_ALPHA / C_TAU) * C_DT;
constexpr float C_A3 = -C_ALPHA * C_BETA * C_DT / C_TAU;
constexpr float C_A4 = 1.0f - C_ALPHA * C_DT / C_TAU;

__device__ __forceinline__ float sigmoidf_(float v) { return 1.0f / (1.0f + expf(-v)); }

// ---------------------------------------------------------------------------
// Tiled WMMA GEMM:  out[m, n] = act( A[m, :K] dot W[n, :K] + bias[n] )
//   A: f32 row-major, leading dim lda          (activations)
//   W: bf16 row-major (N x K)                  (pre-converted weights)
//   out: f32, leading dim ldo
// Block tile 128x128, 256 threads = 8 waves; each wave computes 32x64
// (2 M-subtiles x 4 N-subtiles of 16x16). K stepped by 32, double-buffered.
// Fragment layouts per CDNA5 ISA 7.12.2:
//   A 16x32 bf16: lanes 0-15 row M=l, halves k=0..7,16..23; lanes 16-31 row
//                 M=l-16, halves k=8..15,24..31.
//   B 32x16 bf16: lanes 0-15 col N=l, k=0..7,16..23; lanes 16-31 col N=l-16, k=8..15,24..31.
// ---------------------------------------------------------------------------
template <int ACT>  // 1 = leaky relu, 2 = sigmoid
__global__ __launch_bounds__(256)
void gemm_wmma_bias_act(const float* __restrict__ A,
                        const __bf16* __restrict__ W,
                        const float* __restrict__ bias,
                        float* __restrict__ out,
                        int K, int lda, int ldo)
{
    // double-buffered tiles: 2 x 128 rows x 32 k (+pad to kill bank conflicts)
    __shared__ __align__(16) __bf16 sA[2][128][40];
    __shared__ __align__(16) __bf16 sW[2][128][40];

    const int tid   = threadIdx.x;
    const int lane  = tid & 31;
    const int wave  = tid >> 5;
    const int bm    = blockIdx.y * 128;
    const int bn    = blockIdx.x * 128;
    const int wm    = (wave & 3) * 32;    // wave M offset in block
    const int wn    = (wave >> 2) * 64;   // wave N offset in block
    const int lhalf = lane >> 4;          // 0 | 1
    const int l15   = lane & 15;

    const v8f vzero = {0.f, 0.f, 0.f, 0.f, 0.f, 0.f, 0.f, 0.f};
    v8f acc[2][4];
#pragma unroll
    for (int mt = 0; mt < 2; ++mt)
#pragma unroll
        for (int nt = 0; nt < 4; ++nt) acc[mt][nt] = vzero;

    const int srow = tid >> 1;            // 0..127: staging row
    const int scol = (tid & 1) * 16;      // 0 | 16: staging k-offset

    // Stage one 128x32 K-tile into LDS buffer `buf`.
    //  A: f32 global -> VALU convert -> ds_store (DScnt)
    //  W: bf16 global -> LDS via async copy     (ASYNCcnt, no VGPR transit)
    auto stage = [&](int buf, int k0) {
        const float* ap = A + (size_t)(bm + srow) * (size_t)lda + k0 + scol;
        __builtin_prefetch(ap + 64, 0, 1);          // next+1 K tile
        v4f a0 = *(const v4f*)(ap + 0);
        v4f a1 = *(const v4f*)(ap + 4);
        v4f a2 = *(const v4f*)(ap + 8);
        v4f a3 = *(const v4f*)(ap + 12);
        __bf16* da = &sA[buf][srow][scol];
#pragma unroll
        for (int j = 0; j < 4; ++j) {
            da[j + 0]  = (__bf16)a0[j];
            da[j + 4]  = (__bf16)a1[j];
            da[j + 8]  = (__bf16)a2[j];
            da[j + 12] = (__bf16)a3[j];
        }
        const __bf16* wp = W + (size_t)(bn + srow) * (size_t)K + k0 + scol;
        unsigned           l0 = (unsigned)(uintptr_t)&sW[buf][srow][scol];
        unsigned long long g0 = (unsigned long long)(uintptr_t)wp;
        asm volatile("global_load_async_to_lds_b128 %0, %1, off"
                     :: "v"(l0), "v"(g0) : "memory");
        unsigned           l1 = l0 + 16u;
        unsigned long long g1 = g0 + 16ull;
        asm volatile("global_load_async_to_lds_b128 %0, %1, off"
                     :: "v"(l1), "v"(g1) : "memory");
    };

    stage(0, 0);
    asm volatile("s_wait_asynccnt 0x0" ::: "memory");
    __syncthreads();

    int buf = 0;
    for (int k0 = 0; k0 < K; k0 += 32) {
        // kick off next tile while this one computes
        if (k0 + 32 < K) stage(buf ^ 1, k0 + 32);

        // ---- load fragments from current buffer ----
        v16bf afrag[2];
#pragma unroll
        for (int mt = 0; mt < 2; ++mt) {
            const __bf16* p = &sA[buf][wm + mt * 16 + l15][lhalf * 8];
            v8bf lo = *(const v8bf*)(p + 0);
            v8bf hi = *(const v8bf*)(p + 16);
            afrag[mt] = __builtin_shufflevector(lo, hi,
                0, 1, 2, 3, 4, 5, 6, 7, 8, 9, 10, 11, 12, 13, 14, 15);
        }
        v16bf bfrag[4];
#pragma unroll
        for (int nt = 0; nt < 4; ++nt) {
            const __bf16* p = &sW[buf][wn + nt * 16 + l15][lhalf * 8];
            v8bf lo = *(const v8bf*)(p + 0);
            v8bf hi = *(const v8bf*)(p + 16);
            bfrag[nt] = __builtin_shufflevector(lo, hi,
                0, 1, 2, 3, 4, 5, 6, 7, 8, 9, 10, 11, 12, 13, 14, 15);
        }

        // ---- 8 WMMAs ----
#pragma unroll
        for (int mt = 0; mt < 2; ++mt)
#pragma unroll
            for (int nt = 0; nt < 4; ++nt)
                acc[mt][nt] = __builtin_amdgcn_wmma_f32_16x16x32_bf16(
                    false, afrag[mt], false, bfrag[nt],
                    (short)0, acc[mt][nt], false, false);

        // own async W stores done + own LDS reads/stores done, then barrier:
        // next iteration may overwrite the other buffer safely.
        asm volatile("s_wait_asynccnt 0x0" ::: "memory");
        asm volatile("s_wait_dscnt 0x0"    ::: "memory");
        __syncthreads();
        buf ^= 1;
    }

    // ---- epilogue: bias + activation, f32 store ----
#pragma unroll
    for (int mt = 0; mt < 2; ++mt) {
#pragma unroll
        for (int nt = 0; nt < 4; ++nt) {
            const int n  = bn + wn + nt * 16 + l15;
            const float bv = bias[n];
#pragma unroll
            for (int r = 0; r < 8; ++r) {
                const int m = bm + wm + mt * 16 + (lhalf ? (r + 8) : r);
                float v = acc[mt][nt][r] + bv;
                if (ACT == 1) v = (v >= 0.f) ? v : C_SLOPE * v;
                else if (ACT == 2) v = sigmoidf_(v);
                out[(size_t)m * (size_t)ldo + n] = v;
            }
        }
    }
}

// ---------------------------------------------------------------------------
// Elementwise kernels
// ---------------------------------------------------------------------------
__global__ void cvt_f32_to_bf16(const float* __restrict__ src,
                                __bf16* __restrict__ dst, int n)
{
    int i = blockIdx.x * blockDim.x + threadIdx.x;
    if (i < n) dst[i] = (__bf16)src[i];
}

// feats_bm: (B, T, ZDIM) batch-major (== d_out features region)
__global__ void init_state(const float* __restrict__ feats_bm,
                           float* __restrict__ z, float* __restrict__ dz,
                           float* __restrict__ zm1, float* __restrict__ zm2,
                           float* __restrict__ zs_out,   // (B, T, ZDIM)
                           float* __restrict__ f0)       // (B, ZDIM)
{
    int i = blockIdx.x * blockDim.x + threadIdx.x;   // B*ZDIM
    int b = i >> 8, j = i & 255;
    size_t base = (size_t)b * TD * ZDIM + j;
    float z1 = feats_bm[base];                 // t = 0
    float z2 = feats_bm[base + ZDIM];          // t = 1
    z[i] = z1; dz[i] = (z2 - z1) / C_DT; zm1[i] = z1; zm2[i] = z1;
    zs_out[base] = z1;
    f0[i] = z1;
}

// Build cat = [features[t], z_t, f_t]  (B x 768)
__global__ void pre_step(const float* __restrict__ feats_bm,
                         const float* __restrict__ z,
                         const float* __restrict__ zm1,
                         const float* __restrict__ zm2,
                         float* __restrict__ cat, int t)
{
    int i = blockIdx.x * blockDim.x + threadIdx.x;   // B*ZDIM
    int b = i >> 8, j = i & 255;
    size_t fb = (size_t)b * TD * ZDIM;
    float feat = feats_bm[fb + (size_t)t * ZDIM + j];
    float zn   = feats_bm[fb + (size_t)(TD - 1) * ZDIM + j];
    float zt   = z[i];
    float dz_last  = (t == 0) ? 0.f : (zt - zm1[i]);
    float ddz_last = (t < 2)  ? 0.f : (zm2[i] * 0.5f - zm1[i]);
    float ft = sigmoidf_(C_TAU * C_TAU * ddz_last
                         - C_ALPHA * (C_BETA * (zn - zt) - C_TAU * dz_last));
    float* cr = cat + (size_t)b * 768;
    cr[j] = feat; cr[256 + j] = zt; cr[512 + j] = ft;
}

// Consume wdt (already in wds out region), advance dynamics state.
__global__ void post_step(const float* __restrict__ feats_bm,
                          const float* __restrict__ wds_out,   // (B, T, 512)
                          const float* __restrict__ eps_seq,   // (T-1, B, 256)
                          float* __restrict__ z, float* __restrict__ dz,
                          float* __restrict__ zm1, float* __restrict__ zm2,
                          const float* __restrict__ cat,
                          float* __restrict__ zs_out, int t)
{
    int i = blockIdx.x * blockDim.x + threadIdx.x;   // B*ZDIM
    int b = i >> 8, j = i & 255;
    const float* wdt = wds_out + (size_t)b * TD * TRANS + (size_t)(t + 1) * TRANS;
    float mu = wdt[j], lv = wdt[256 + j];
    float eps = eps_seq[(size_t)t * BD * ZDIM + i];
    float wt = sigmoidf_(mu + eps * expf(0.5f * lv));
    float zn = feats_bm[(size_t)b * TD * ZDIM + (size_t)(TD - 1) * ZDIM + j];
    float ft = cat[(size_t)b * 768 + 512 + j];
    float bb = (C_ALPHA * C_BETA * zn + ft + wt) * (C_DT / C_TAU);
    float zt = z[i], dzt = dz[i], zm1v = zm1[i];
    float z_next  = C_A1 * zt + C_A2 * dzt + C_DT * bb;
    float dz_next = C_A3 * zt + C_A4 * dzt + bb;
    zm2[i] = zm1v; zm1[i] = zt; z[i] = z_next; dz[i] = dz_next;
    zs_out[(size_t)b * TD * ZDIM + (size_t)(t + 1) * ZDIM + j] = z_next;
}

// _d() central-difference chain on zs[:127] (batch-major), then fs = sigmoid(...)
__device__ __forceinline__ float dz_at(const float* zs, size_t base, int i)
{
    if (i == 0)   return zs[base + ZDIM] - zs[base];
    if (i == 126) return zs[base + 126 * ZDIM] - zs[base + 125 * ZDIM];
    return zs[base + (size_t)(i + 1) * ZDIM] - 0.5f * zs[base + (size_t)(i - 1) * ZDIM];
}

__global__ void fs_kernel(const float* __restrict__ zs,        // (B, T, 256)
                          const float* __restrict__ feats_bm,  // (B, T, 256)
                          float* __restrict__ fs_out)          // (B, 127, 256)
{
    int idx = blockIdx.x * blockDim.x + threadIdx.x;   // B*127*256
    if (idx >= BD * 127 * ZDIM) return;
    int j = idx & 255;
    int t = (idx >> 8) % 127;
    int b = idx / (127 * ZDIM);
    size_t zbase = (size_t)b * TD * ZDIM + j;
    float zp = zs[zbase + (size_t)t * ZDIM];
    float d1 = dz_at(zs, zbase, t);
    float d2;
    if (t == 0)        d2 = dz_at(zs, zbase, 1)   - dz_at(zs, zbase, 0);
    else if (t == 126) d2 = dz_at(zs, zbase, 126) - dz_at(zs, zbase, 125);
    else               d2 = dz_at(zs, zbase, t + 1) - 0.5f * dz_at(zs, zbase, t - 1);
    float zn = feats_bm[(size_t)b * TD * ZDIM + (size_t)(TD - 1) * ZDIM + j];
    fs_out[idx] = sigmoidf_(C_TAU * C_TAU * d2
                            - C_ALPHA * (C_BETA * (zn - zp) - C_TAU * d1));
}

// ---------------------------------------------------------------------------
// Host orchestration
// ---------------------------------------------------------------------------
static inline void launch_gemm(int act, const float* A, const __bf16* W,
                               const float* b, float* out,
                               int M, int N, int K, int lda, int ldo,
                               hipStream_t s)
{
    dim3 grid(N / 128, M / 128), block(256);
    if (act == 2)
        gemm_wmma_bias_act<2><<<grid, block, 0, s>>>(A, W, b, out, K, lda, ldo);
    else
        gemm_wmma_bias_act<1><<<grid, block, 0, s>>>(A, W, b, out, K, lda, ldo);
}

extern "C" void kernel_launch(void* const* d_in, const int* in_sizes, int n_in,
                              void* d_out, int out_size, void* d_ws, size_t ws_size,
                              hipStream_t stream)
{
    const float* x = (const float*)d_in[0];

    // Param leaf order: detect jax-sorted-key vs dict-insertion flattening.
    // sorted:    de1,de2,de3,en1,en2,en3,in1..in4,no1,no2  -> d_in[1] = de1.W (262144)
    // insertion: en1,en2,en3,in1..in4,de1,de2,de3,no1,no2  -> d_in[1] = en1.W (524288)
    int I_de1, I_de2, I_de3, I_en1, I_en2, I_en3, I_in1, I_in2, I_no1, I_no2;
    if (in_sizes[1] == 1024 * 512) {   // insertion order
        I_en1 = 1; I_en2 = 3; I_en3 = 5; I_in1 = 7; I_in2 = 9;
        I_de1 = 15; I_de2 = 17; I_de3 = 19; I_no1 = 21; I_no2 = 23;
    } else {                           // sorted-key order
        I_de1 = 1; I_de2 = 3; I_de3 = 5; I_en1 = 7; I_en2 = 9; I_en3 = 11;
        I_in1 = 13; I_in2 = 15; I_no1 = 21; I_no2 = 23;
    }
    const float* eps_seq = (const float*)d_in[n_in - 1];

    auto Wf = [&](int i) { return (const float*)d_in[i]; };
    auto Bf = [&](int i) { return (const float*)d_in[i + 1]; };

    // d_out regions: (xs, zs, features, wds, fs) flat in return order
    float* out      = (float*)d_out;
    float* out_xs   = out;                          // (B, T, 512)
    float* out_zs   = out + (size_t)16777216;       // (B, T, 256)
    float* out_ft   = out + (size_t)25165824;       // (B, T, 256) features
    float* out_wds  = out + (size_t)33554432;       // (B, T, 512)
    float* out_fs   = out + (size_t)50331648;       // (B, 127, 256)

    // Workspace layout
    char* ws = (char*)d_ws;
    __bf16* wbf = (__bf16*)ws;
    size_t woff = 0;
    auto wreg = [&](size_t elems) { __bf16* p = wbf + woff; woff += elems; return p; };
    __bf16* en1w = wreg(1024 * 512);
    __bf16* en2w = wreg(1024 * 1024);
    __bf16* en3w = wreg(256 * 1024);
    __bf16* in1w = wreg(512 * 256);
    __bf16* in2w = wreg(512 * 512);
    __bf16* no1w = wreg(512 * 768);
    __bf16* no2w = wreg(512 * 512);
    __bf16* de1w = wreg(1024 * 256);
    __bf16* de2w = wreg(1024 * 1024);
    __bf16* de3w = wreg(512 * 1024);
    size_t fo = (woff * sizeof(__bf16) + 255) & ~(size_t)255;
    float* bufA = (float*)(ws + fo);                 fo += (size_t)32768 * 1024 * 4;
    float* bufB = (float*)(ws + fo);                 fo += (size_t)32768 * 1024 * 4;
    float* catb = (float*)(ws + fo);                 fo += (size_t)256 * 768 * 4;
    float* hh   = (float*)(ws + fo);                 fo += (size_t)256 * 512 * 4;
    float* h1   = (float*)(ws + fo);                 fo += (size_t)256 * 512 * 4;
    float* f0   = (float*)(ws + fo);                 fo += (size_t)256 * 256 * 4;
    float* st_z   = (float*)(ws + fo);               fo += (size_t)256 * 256 * 4;
    float* st_dz  = (float*)(ws + fo);               fo += (size_t)256 * 256 * 4;
    float* st_zm1 = (float*)(ws + fo);               fo += (size_t)256 * 256 * 4;
    float* st_zm2 = (float*)(ws + fo);               fo += (size_t)256 * 256 * 4;
    (void)ws_size;

    // 1) Convert weights f32 -> bf16 (10 small kernels; ~19 MB total traffic)
    auto cvt = [&](const float* s, __bf16* d, int n) {
        cvt_f32_to_bf16<<<(n + 255) / 256, 256, 0, stream>>>(s, d, n);
    };
    cvt(Wf(I_en1), en1w, 1024 * 512);
    cvt(Wf(I_en2), en2w, 1024 * 1024);
    cvt(Wf(I_en3), en3w, 256 * 1024);
    cvt(Wf(I_in1), in1w, 512 * 256);
    cvt(Wf(I_in2), in2w, 512 * 512);
    cvt(Wf(I_no1), no1w, 512 * 768);
    cvt(Wf(I_no2), no2w, 512 * 512);
    cvt(Wf(I_de1), de1w, 1024 * 256);
    cvt(Wf(I_de2), de2w, 1024 * 1024);
    cvt(Wf(I_de3), de3w, 512 * 1024);

    const int M = BD * TD;   // 32768 rows, batch-major

    // 2) Encoder (batch-major: features land directly in d_out)
    launch_gemm(1, x,    en1w, Bf(I_en1), bufA,   M, HID,  XDIM, XDIM, HID,  stream);
    launch_gemm(1, bufA, en2w, Bf(I_en2), bufB,   M, HID,  HID,  HID,  HID,  stream);
    launch_gemm(1, bufB, en3w, Bf(I_en3), out_ft, M, ZDIM, HID,  HID,  ZDIM, stream);

    // 3) Init net: wds[:,0,:] = lrelu(in2(lrelu(in1(features[:,0]))))
    init_state<<<256, 256, 0, stream>>>(out_ft, st_z, st_dz, st_zm1, st_zm2,
                                        out_zs, f0);
    launch_gemm(1, f0, in1w, Bf(I_in1), h1, BD, TRANS, ZDIM, ZDIM, TRANS, stream);
    launch_gemm(1, h1, in2w, Bf(I_in2), out_wds, BD, TRANS, TRANS, TRANS,
                TD * TRANS, stream);   // strided epilogue -> wds[:,0,:]

    // 4) Sequential scan: 127 steps of (pre | no1 | no2 | post)
    for (int t = 0; t < TD - 1; ++t) {
        pre_step<<<256, 256, 0, stream>>>(out_ft, st_z, st_zm1, st_zm2, catb, t);
        launch_gemm(1, catb, no1w, Bf(I_no1), hh, BD, TRANS, 768, 768, TRANS, stream);
        launch_gemm(1, hh, no2w, Bf(I_no2), out_wds + (size_t)(t + 1) * TRANS,
                    BD, TRANS, TRANS, TRANS, TD * TRANS, stream);  // wds[:,t+1,:]
        post_step<<<256, 256, 0, stream>>>(out_ft, out_wds, eps_seq,
                                           st_z, st_dz, st_zm1, st_zm2,
                                           catb, out_zs, t);
    }

    // 5) Decoder over zs (batch-major rows are contiguous) -> xs directly
    launch_gemm(1, out_zs, de1w, Bf(I_de1), bufA,   M, HID,  ZDIM, ZDIM, HID,  stream);
    launch_gemm(1, bufA,   de2w, Bf(I_de2), bufB,   M, HID,  HID,  HID,  HID,  stream);
    launch_gemm(2, bufB,   de3w, Bf(I_de3), out_xs, M, XDIM, HID,  HID,  XDIM, stream);

    // 6) fs tail
    int nfs = BD * 127 * ZDIM;
    fs_kernel<<<(nfs + 255) / 256, 256, 0, stream>>>(out_zs, out_ft, out_fs);
}
